// CircuitRankNet_47983374631310
// MI455X (gfx1250) — matrix operand order repaired and checked by reading, not compiled, not run on hardware
//
#include <hip/hip_runtime.h>
#include <hip/hip_bf16.h>

// ---------------------------------------------------------------------------
// Problem constants (fixed by the reference)
#define NN 100000
#define EE 1600000
#define FF 64
#define GG 64

// ---------------------------------------------------------------------------
// WMMA vector types (CDNA5 gfx1250, wave32)
typedef __attribute__((ext_vector_type(16))) __bf16 v16bf;
typedef __attribute__((ext_vector_type(8)))  float  v8f;

union BF16x16 {
    unsigned short u[16];
    uint4          q[2];
    v16bf          v;
};

__device__ __forceinline__ unsigned short f2bf(float f) {
    // round-to-nearest-even f32 -> bf16 (used only in the one-shot pack kernel)
    unsigned int u = __float_as_uint(f);
    u += 0x7FFFu + ((u >> 16) & 1u);
    return (unsigned short)(u >> 16);
}

// ---------------------------------------------------------------------------
// Degree count: cnt[dst[e]] += 1
__global__ __launch_bounds__(256) void deg_kernel(const int* __restrict__ dst,
                                                  float* __restrict__ cnt, int E) {
    int e = blockIdx.x * blockDim.x + threadIdx.x;
    if (e < E) atomicAdd(&cnt[dst[e]], 1.0f);
}

// Nodes per graph: gcnt[batch[i]] += 1
__global__ __launch_bounds__(256) void gcnt_kernel(const int* __restrict__ batch,
                                                   float* __restrict__ gcnt, int n) {
    int i = blockIdx.x * blockDim.x + threadIdx.x;
    if (i < n) atomicAdd(&gcnt[batch[i]], 1.0f);
}

// ---------------------------------------------------------------------------
// One-shot weight repack: [wl ; wr] (2*KHALF x 128, f32 row-major) -> bf16 in
// exact WMMA B-fragment order:
//   out[(((kt*8 + colTile)*32 + lane)*16 + i]  holds
//     B[kt*32 + (lane>>4)*8 + (i&7) + (i>>3)*16][colTile*16 + (lane&15)]
// so the GEMM reads a B fragment as two contiguous b128 loads.
__global__ __launch_bounds__(256) void pack_w_kernel(
    const float* __restrict__ wl, const float* __restrict__ wr,
    unsigned short* __restrict__ out, int KHALF) {
    int id = blockIdx.x * blockDim.x + threadIdx.x;
    int total = (2 * KHALF / 32) * 8 * 32 * 16;
    if (id >= total) return;
    int i    = id & 15;
    int lane = (id >> 4) & 31;
    int ct   = (id >> 9) & 7;
    int kt   = id >> 12;
    int k = kt * 32 + (lane >> 4) * 8 + (i & 7) + (i >> 3) * 16;
    int n = ct * 16 + (lane & 15);
    float w = (k < KHALF) ? wl[k * 128 + n] : wr[(k - KHALF) * 128 + n];
    out[id] = f2bf(w);
}

// ---------------------------------------------------------------------------
// Edge gather + scatter-add: agg[dst] += x[src], vectorized 4 floats/thread.
// SHIFT = log2(D/4). Dominant memory phase -> float4 loads, f32 atomics.
template <int SHIFT>
__global__ __launch_bounds__(256) void scatter_add_kernel(
    const float* __restrict__ x, const int* __restrict__ src,
    const int* __restrict__ dst, float* __restrict__ agg,
    long long total, int D) {
    long long tid = (long long)blockIdx.x * blockDim.x + threadIdx.x;
    if (tid >= total) return;
    long long e  = tid >> SHIFT;
    int       f4 = (int)(tid & ((1 << SHIFT) - 1)) << 2;
    int s = src[e], d = dst[e];
    float4 v = *(const float4*)(x + (long long)s * D + f4);
    float* o = agg + (long long)d * D + f4;
    atomicAdd(o + 0, v.x);
    atomicAdd(o + 1, v.y);
    atomicAdd(o + 2, v.z);
    atomicAdd(o + 3, v.w);
}

// ---------------------------------------------------------------------------
// Fused SAGE linear:  out = [agg/max(cnt,1) | xin] @ [wl ; wr] + bias
// KHALF = feature width of each half (64 for layer1, 128 for layer2).
// Output width is fixed at 128. One 16x16 tile per wave via
// v_wmma_f32_16x16x32_bf16 (bf16 A/B, f32 accumulate). Weights arrive
// pre-packed in fragment order (wpack). Degree normalization and f32->bf16
// conversion (native v_cvt, via __bf16 casts) are fused into the A load.
// POOL=true: instead of storing h2, atomically mean-pool rows into
// pooled[batch[row]*128 + col] (divide by graph size happens later).
template <int KHALF, bool POOL>
__global__ __launch_bounds__(256) void sage_gemm_kernel(
    const float* __restrict__ agg,            // [N, KHALF]
    const float* __restrict__ xin,            // [N, KHALF]
    const float* __restrict__ cnt,            // [N]
    const unsigned short* __restrict__ wpack, // packed bf16 [ktiles][8][32][16]
    const float* __restrict__ bias,           // [128]
    float* __restrict__ out,                  // [N, 128]  (if !POOL)
    const int* __restrict__ batch,            // [N]       (if POOL)
    float* __restrict__ pooled)               // [G, 128]  (if POOL)
{
    const int wave = threadIdx.x >> 5;
    const int lane = threadIdx.x & 31;
    const int tile = blockIdx.x * 8 + wave;   // 8 waves per block
    const int tileM = tile >> 3;              // 8 column tiles per row group
    const int tileN = tile & 7;
    const int row0 = tileM * 16;
    const int col0 = tileN * 16;
    const int m    = lane & 15;
    const int half = lane >> 4;               // 0: lanes 0-15, 1: lanes 16-31
    const int row  = row0 + m;
    const int colg = col0 + m;

    // degree normalization fused into the A-fragment load
    const float sc = 1.0f / fmaxf(cnt[row], 1.0f);

    v8f c = {};  // f32 accumulator, 16x16 tile

    const int KT = 2 * KHALF;
#pragma unroll
    for (int kb = 0; kb < KT; kb += 32) {
        // ---- A fragment: 16x32 bf16 (row = m, K-subchunks per lane-half) ----
        const float* arow;
        int kcol;
        float s;
        if (kb < KHALF) { arow = agg + (long long)row * KHALF; kcol = kb;         s = sc;   }
        else            { arow = xin + (long long)row * KHALF; kcol = kb - KHALF; s = 1.0f; }
        const float4* pa0 = (const float4*)(arow + kcol + half * 8);
        const float4* pa1 = (const float4*)(arow + kcol + 16 + half * 8);
        float4 q0 = pa0[0], q1 = pa0[1];
        float4 q2 = pa1[0], q3 = pa1[1];
        v16bf A;
        A[0]  = (__bf16)(q0.x * s); A[1]  = (__bf16)(q0.y * s);
        A[2]  = (__bf16)(q0.z * s); A[3]  = (__bf16)(q0.w * s);
        A[4]  = (__bf16)(q1.x * s); A[5]  = (__bf16)(q1.y * s);
        A[6]  = (__bf16)(q1.z * s); A[7]  = (__bf16)(q1.w * s);
        A[8]  = (__bf16)(q2.x * s); A[9]  = (__bf16)(q2.y * s);
        A[10] = (__bf16)(q2.z * s); A[11] = (__bf16)(q2.w * s);
        A[12] = (__bf16)(q3.x * s); A[13] = (__bf16)(q3.y * s);
        A[14] = (__bf16)(q3.z * s); A[15] = (__bf16)(q3.w * s);

        // ---- B fragment: pre-packed bf16, two contiguous b128 loads ----
        const unsigned short* wp =
            wpack + (size_t)((((kb >> 5) * 8 + tileN) * 32 + lane) << 4);
        BF16x16 B;
        B.q[0] = *(const uint4*)(wp);
        B.q[1] = *(const uint4*)(wp + 8);

        c = __builtin_amdgcn_wmma_f32_16x16x32_bf16(
            /*neg_a=*/false, A, /*neg_b=*/false, B.v,
            /*c_mod=*/(short)0, c, /*reuse_a=*/false, /*reuse_b=*/false);
    }

    const float bv = bias[colg];
    if (!POOL) {
#pragma unroll
        for (int r = 0; r < 8; ++r)
            out[(long long)(row0 + half * 8 + r) * 128 + colg] = c[r] + bv;
    } else {
#pragma unroll
        for (int r = 0; r < 8; ++r) {
            int rr = row0 + half * 8 + r;
            atomicAdd(&pooled[(long long)batch[rr] * 128 + colg], c[r] + bv);
        }
    }
}

// ---------------------------------------------------------------------------
// Comparator MLP on pooled embeddings. One block (128 threads) per graph.
// f = [pooled0/n0 | pooled1/n1] (256) -> sigmoid(f@cw1+cb1) (128)
//   -> sigmoid(h@cw2+cb2) -> out[g]
__global__ __launch_bounds__(128) void comparator_kernel(
    const float* __restrict__ pooled0, const float* __restrict__ pooled1,
    const float* __restrict__ gcnt0,   const float* __restrict__ gcnt1,
    const float* __restrict__ cw1, const float* __restrict__ cb1,
    const float* __restrict__ cw2, const float* __restrict__ cb2,
    float* __restrict__ outp) {
    __shared__ float fbuf[256];
    __shared__ float hbuf[128];
    const int g = blockIdx.x;
    const int t = threadIdx.x;
    const float inv0 = 1.0f / fmaxf(gcnt0[g], 1.0f);
    const float inv1 = 1.0f / fmaxf(gcnt1[g], 1.0f);
    fbuf[t]       = pooled0[g * 128 + t] * inv0;
    fbuf[t + 128] = pooled1[g * 128 + t] * inv1;
    __syncthreads();
    float acc = cb1[t];
#pragma unroll 8
    for (int k = 0; k < 256; ++k) acc += fbuf[k] * cw1[k * 128 + t];
    float hv = 1.0f / (1.0f + __expf(-acc));
    hbuf[t] = hv * cw2[t];
    __syncthreads();
    for (int s = 64; s > 0; s >>= 1) {
        if (t < s) hbuf[t] += hbuf[t + s];
        __syncthreads();
    }
    if (t == 0) outp[g] = 1.0f / (1.0f + __expf(-(hbuf[0] + cb2[0])));
}

// ---------------------------------------------------------------------------
extern "C" void kernel_launch(void* const* d_in, const int* in_sizes, int n_in,
                              void* d_out, int out_size, void* d_ws, size_t ws_size,
                              hipStream_t stream) {
    const float* x0  = (const float*)d_in[0];
    const int*   ei0 = (const int*)d_in[1];
    const int*   b0  = (const int*)d_in[2];
    const float* x1  = (const float*)d_in[3];
    const int*   ei1 = (const int*)d_in[4];
    const int*   b1  = (const int*)d_in[5];
    const float* w1l = (const float*)d_in[6];
    const float* b1l = (const float*)d_in[7];
    const float* w1r = (const float*)d_in[8];
    const float* w2l = (const float*)d_in[9];
    const float* b2l = (const float*)d_in[10];
    const float* w2r = (const float*)d_in[11];
    const float* cw1 = (const float*)d_in[12];
    const float* cb1 = (const float*)d_in[13];
    const float* cw2 = (const float*)d_in[14];
    const float* cb2 = (const float*)d_in[15];

    // Workspace carve-out (~103 MB), reused across the two graphs.
    float* agg     = (float*)d_ws;                  // N*128
    float* h1      = agg + (size_t)NN * 128;        // N*128
    float* cnt     = h1 + (size_t)NN * 128;         // N
    float* pooled0 = cnt + NN;                      // G*128
    float* pooled1 = pooled0 + GG * 128;            // G*128
    float* gcnt0   = pooled1 + GG * 128;            // G
    float* gcnt1   = gcnt0 + GG;                    // G
    // packed bf16 weights (16-byte aligned: offset so far is a multiple of 16B)
    unsigned short* wpack1 = (unsigned short*)(gcnt1 + GG);   // 4*8*32*16 = 16384
    unsigned short* wpack2 = wpack1 + 16384;                  // 8*8*32*16 = 32768

    // One-shot weight repack into WMMA B-fragment order (bf16).
    pack_w_kernel<<<(16384 + 255) / 256, 256, 0, stream>>>(w1l, w1r, wpack1, 64);
    pack_w_kernel<<<(32768 + 255) / 256, 256, 0, stream>>>(w2l, w2r, wpack2, 128);

    const int gemm_blocks = (NN / 16) * 8 / 8;      // 6250: 1 tile/wave, 8 waves/block

    for (int g = 0; g < 2; ++g) {
        const float* x    = g ? x1 : x0;
        const int*   src  = g ? ei1 : ei0;          // edge_index row 0
        const int*   dstE = src + EE;               // edge_index row 1
        const int*   bat  = g ? b1 : b0;
        float* pooled = g ? pooled1 : pooled0;
        float* gcnt   = g ? gcnt1 : gcnt0;

        hipMemsetAsync(cnt, 0, NN * sizeof(float), stream);
        hipMemsetAsync(agg, 0, (size_t)NN * 64 * sizeof(float), stream);
        hipMemsetAsync(pooled, 0, GG * 128 * sizeof(float), stream);
        hipMemsetAsync(gcnt, 0, GG * sizeof(float), stream);

        deg_kernel<<<(EE + 255) / 256, 256, 0, stream>>>(dstE, cnt, EE);
        gcnt_kernel<<<(NN + 255) / 256, 256, 0, stream>>>(bat, gcnt, NN);

        // SAGE1 aggregate: E x 16 float4-threads
        {
            long long total = (long long)EE * 16;
            int blocks = (int)((total + 255) / 256);
            scatter_add_kernel<4><<<blocks, 256, 0, stream>>>(x, src, dstE, agg, total, 64);
        }
        // h1 = [agg/cnt | x] @ [w1l ; w1r] + b1l     (K = 128, bf16 WMMA)
        sage_gemm_kernel<64, false><<<gemm_blocks, 256, 0, stream>>>(
            agg, x, cnt, wpack1, b1l, h1, nullptr, nullptr);

        hipMemsetAsync(agg, 0, (size_t)NN * 128 * sizeof(float), stream);
        // SAGE2 aggregate: E x 32 float4-threads
        {
            long long total = (long long)EE * 32;
            int blocks = (int)((total + 255) / 256);
            scatter_add_kernel<5><<<blocks, 256, 0, stream>>>(h1, src, dstE, agg, total, 128);
        }
        // h2 = [agg/cnt | h1] @ [w2l ; w2r] + b2l, fused mean-pool into pooled
        sage_gemm_kernel<128, true><<<gemm_blocks, 256, 0, stream>>>(
            agg, h1, cnt, wpack2, b2l, nullptr, bat, pooled);
    }

    comparator_kernel<<<GG, 128, 0, stream>>>(pooled0, pooled1, gcnt0, gcnt1,
                                              cw1, cb1, cw2, cb2, (float*)d_out);
}
